// InterlacedPoolAttention_76725295776122
// MI455X (gfx1250) — compile-verified, matrix operand order checked by reading, not run on hardware
//
#include <hip/hip_runtime.h>
#include <hip/hip_bf16.h>

// ---------------------------------------------------------------------------
// CDNA5 (gfx1250) windowed attention:  QKV gemm -> per-window attention -> proj
// Matmuls via v_wmma_f32_16x16x32_f16 (wave32); weight tiles staged in LDS via
// async global->LDS loads (double buffered), software-pipelined k-loops.
// ---------------------------------------------------------------------------

typedef __attribute__((ext_vector_type(16))) _Float16 v16h;
typedef __attribute__((ext_vector_type(8)))  _Float16 v8h;
typedef __attribute__((ext_vector_type(8)))  float    v8f;
typedef __attribute__((ext_vector_type(4)))  int      v4i;

#define DEV __device__ __forceinline__

static constexpr int WSZ   = 7;
static constexpr int LQ    = 49;     // window tokens
static constexpr int LP    = 64;     // padded to multiple of 16
static constexpr int NHEAD = 16;
static constexpr int HD    = 32;     // head dim
static constexpr int CH    = 512;
static constexpr int QH    = 19;     // ceil(133/7)
static constexpr int QW    = 19;
static constexpr int NB    = 8 * QH * QW;   // 2888 windows
static constexpr int IMG   = 128;
static constexpr int PADT  = 2;      // ph//2 (ph = 133-128 = 5)
static constexpr float QSCALE = 0.17677669529663689f;  // 32^-0.5

// workspace element counts (halfs)
static constexpr size_t N_WQKV = 1536ull * 512;
static constexpr size_t N_WOUT = 512ull * 512;
static constexpr size_t N_QKV1 = (size_t)NB * NHEAD * LP * HD;

// --- async global->LDS support (guarded; falls back to reg-staged copy) -----

#if __has_builtin(__builtin_amdgcn_global_load_async_to_lds_b128)
#define HAVE_ASYNC_LDS 1
#else
#define HAVE_ASYNC_LDS 0
#endif

#if HAVE_ASYNC_LDS
typedef __attribute__((address_space(1))) v4i gv4i;   // global int4
typedef __attribute__((address_space(3))) v4i lv4i;   // LDS int4
#endif

DEV void copy16_g2l(const _Float16* g, _Float16* l) {
#if HAVE_ASYNC_LDS
  __builtin_amdgcn_global_load_async_to_lds_b128((gv4i*)g, (lv4i*)l, 0, 0);
#else
  *(v8h*)l = *(const v8h*)g;
#endif
}

DEV void wait_async() {
#if HAVE_ASYNC_LDS
#if __has_builtin(__builtin_amdgcn_s_wait_asynccnt)
  __builtin_amdgcn_s_wait_asynccnt(0);
#else
  asm volatile("s_wait_asynccnt 0x0" ::: "memory");
#endif
#endif
}

// --- WMMA helpers -----------------------------------------------------------

DEV v8f wmma16(v16h a, v16h b, v8f c) {
  return __builtin_amdgcn_wmma_f32_16x16x32_f16(
      /*neg_a=*/false, a, /*neg_b=*/false, b,
      /*c_mod=*/(short)0, c, /*reuse_a=*/false, /*reuse_b=*/false);
}

// Load a 16x32 f16 fragment (A row-major [m*ld+k], or B when element (k,n)
// lives at p[n*ld+k]).  Works on global or LDS pointers.
DEV v16h load_frag16(const _Float16* p, int ld) {
  int lane = threadIdx.x & 31;
  int lr = lane & 15;          // row (A) / column (B)
  int lg = lane >> 4;          // k-group select
  const _Float16* q = p + (size_t)lr * ld + lg * 8;
  v8h c0 = *(const v8h*)q;          // K = lg*8 .. lg*8+7
  v8h c1 = *(const v8h*)(q + 16);   // K = lg*8+16 .. lg*8+23
  return __builtin_shufflevector(c0, c1, 0,1,2,3,4,5,6,7,8,9,10,11,12,13,14,15);
}

// A fragment gathered straight from padded window rows of x (f32 -> f16).
DEV v16h load_frag_from_x(const float* xrow, int kk, int lg) {
  v16h r;
  if (xrow) {
    const float* q = xrow + kk + lg * 8;
#pragma unroll
    for (int i = 0; i < 8; ++i) r[i]     = (_Float16)q[i];
#pragma unroll
    for (int i = 0; i < 8; ++i) r[8 + i] = (_Float16)q[16 + i];
  } else {
#pragma unroll
    for (int i = 0; i < 16; ++i) r[i] = (_Float16)0.0f;
  }
  return r;
}

DEV void decode_window(int w, int& b, int& qy, int& qx) {
  b = w / (QH * QW);
  int rem = w - b * (QH * QW);
  qy = rem / QW;
  qx = rem - qy * QW;
}

// --- Kernel 0: weights f32 -> f16 ------------------------------------------

__global__ void cvt_weights_kernel(const float* __restrict__ wqkv,
                                   const float* __restrict__ wout,
                                   _Float16* __restrict__ wqkv16,
                                   _Float16* __restrict__ wout16) {
  int stride = gridDim.x * blockDim.x;
  for (size_t i = blockIdx.x * blockDim.x + threadIdx.x; i < N_WQKV; i += stride)
    wqkv16[i] = (_Float16)wqkv[i];
  for (size_t i = blockIdx.x * blockDim.x + threadIdx.x; i < N_WOUT; i += stride)
    wout16[i] = (_Float16)wout[i];
}

// --- Kernel 1: QKV projection (window-gather fused into A loads) ------------
// grid (NB, 12), block 256 (8 waves).  Block tile: 64 M x 128 N.
// B (weights) staged in LDS, double-buffered via async global->LDS copies.

__global__ void __launch_bounds__(256)
qkv_gemm_kernel(const float* __restrict__ x,
                const _Float16* __restrict__ wqkv16,
                const float* __restrict__ bqkv,
                _Float16* __restrict__ qbuf,
                _Float16* __restrict__ kbuf,
                _Float16* __restrict__ vbufT) {
  __shared__ _Float16 Bs[2][128 * 32];   // 16 KB, double buffered

  const int w = blockIdx.x;
  int b, qy, qx;
  decode_window(w, b, qy, qx);

  const int wid = threadIdx.x >> 5;
  const int lane = threadIdx.x & 31;
  const int lr = lane & 15, lg = lane >> 4;
  const int m0 = (wid & 3) * 16;
  const int nh = wid >> 2;               // 0/1: n-half within the 128 block
  const int nb0 = blockIdx.y * 128;

  // per-lane source row pointer for A fragments (zero for pad rows/pixels)
  const float* xrow = nullptr;
  {
    int l = m0 + lr;
    if (l < LQ) {
      int iy = l / WSZ, ix = l - iy * WSZ;
      int y  = qy * WSZ + iy - PADT;
      int xc = qx * WSZ + ix - PADT;
      if ((unsigned)y < (unsigned)IMG && (unsigned)xc < (unsigned)IMG)
        xrow = x + ((size_t)b * (IMG * IMG) + (size_t)y * IMG + xc) * CH;
    }
  }

  // cooperative B tile copy: 128 cols x 32 halfs = 512 x 16B chunks
  auto stage_B = [&](int kk, int buf) {
#pragma unroll
    for (int c = 0; c < 2; ++c) {
      int ci = c * 256 + threadIdx.x;
      int np = ci >> 2;
      int ho = (ci & 3) * 8;
      copy16_g2l(wqkv16 + (size_t)(nb0 + np) * CH + kk + ho,
                 &Bs[buf][np * 32 + ho]);
    }
  };

  v8f acc[4];
  const v8f vzero = {0.f,0.f,0.f,0.f,0.f,0.f,0.f,0.f};
#pragma unroll
  for (int j = 0; j < 4; ++j) acc[j] = vzero;

  // pipelined k-loop
  stage_B(0, 0);
  v16h afCur = load_frag_from_x(xrow, 0, lg);
  wait_async();
  __syncthreads();

  int buf = 0;
  for (int kk = 0; kk < CH; kk += 32) {
    const int kn = kk + 32;
    const bool hasNext = kn < CH;
    if (hasNext) stage_B(kn, buf ^ 1);
    v16h afNext = afCur;
    if (hasNext) afNext = load_frag_from_x(xrow, kn, lg);

#pragma unroll
    for (int j = 0; j < 4; ++j) {
      v16h bf = load_frag16(&Bs[buf][(nh * 64 + j * 16) * 32], 32);
      acc[j] = wmma16(afCur, bf, acc[j]);
    }

    wait_async();
    __syncthreads();
    afCur = afNext;
    buf ^= 1;
  }

  // epilogue: + bias, split into q (scaled) / k / v^T
#pragma unroll
  for (int j = 0; j < 4; ++j) {
    int n = nb0 + nh * 64 + j * 16 + lr;
    float bn = bqkv[n];
    int region = n >> 9;            // 0=q 1=k 2=v (uniform per j-tile)
    int nn = n & 511;
    int h = nn >> 5, d = nn & 31;
    size_t whb = ((size_t)w * NHEAD + h);
#pragma unroll
    for (int r = 0; r < 8; ++r) {
      int l = m0 + r + 8 * lg;
      float val = acc[j][r] + bn;
      if (region == 0) {
        qbuf[(whb * LP + l) * HD + d] = (_Float16)(val * QSCALE);
      } else if (region == 1) {
        kbuf[(whb * LP + l) * HD + d] = (_Float16)val;
      } else {
        vbufT[(whb * HD + d) * LP + l] = (_Float16)val;   // transposed
      }
    }
  }
}

// --- Kernel 2: per-(window,head) attention, one wave each -------------------
// S = q k^T + rpe ; masked softmax over 49 cols ; O = P v.  P staged in LDS.

__global__ void __launch_bounds__(256)
attn_kernel(const _Float16* __restrict__ qbuf,
            const _Float16* __restrict__ kbuf,
            const _Float16* __restrict__ vbufT,
            const float* __restrict__ rpe,
            _Float16* __restrict__ abuf) {
  __shared__ _Float16 Plds[8 * LP * LP];   // 64 KB, 8 KB per wave

  const int wid = threadIdx.x >> 5;
  const int lane = threadIdx.x & 31;
  const int lr = lane & 15, lg = lane >> 4;

  const int gid = blockIdx.x * 8 + wid;    // (window, head) pair
  const int w = gid >> 4;
  const int h = gid & 15;

  const _Float16* qb = qbuf  + ((size_t)w * NHEAD + h) * (LP * HD);
  const _Float16* kb = kbuf  + ((size_t)w * NHEAD + h) * (LP * HD);
  const _Float16* vt = vbufT + ((size_t)w * NHEAD + h) * (HD * LP);

  // ---- S = q k^T  (K = 32 in one WMMA step per tile) ----
  v16h qf[4];
#pragma unroll
  for (int mt = 0; mt < 4; ++mt) qf[mt] = load_frag16(qb + (size_t)(mt * 16) * HD, HD);

  v8f S[4][4];
  const v8f vzero = {0.f,0.f,0.f,0.f,0.f,0.f,0.f,0.f};
#pragma unroll
  for (int mt = 0; mt < 4; ++mt)
#pragma unroll
    for (int nt = 0; nt < 4; ++nt) S[mt][nt] = vzero;

#pragma unroll
  for (int nt = 0; nt < 4; ++nt) {
    v16h kf = load_frag16(kb + (size_t)(nt * 16) * HD, HD);
#pragma unroll
    for (int mt = 0; mt < 4; ++mt) S[mt][nt] = wmma16(qf[mt], kf, S[mt][nt]);
  }

  // ---- + relative position bias, mask cols >= 49, row softmax ----
#pragma unroll
  for (int mt = 0; mt < 4; ++mt) {
#pragma unroll
    for (int r = 0; r < 8; ++r) {
      int l = mt * 16 + r + 8 * lg;          // query index (row)
      int ly = l / WSZ, lx = l - ly * WSZ;
      bool lvalid = (l < LQ);
#pragma unroll
      for (int nt = 0; nt < 4; ++nt) {
        int m = nt * 16 + lr;                // key index (col)
        if (m >= LQ) {
          S[mt][nt][r] = -1e30f;
        } else if (lvalid) {
          int my = m / WSZ, mx = m - my * WSZ;
          int ridx = (ly - my + WSZ - 1) * (2 * WSZ - 1) + (lx - mx + WSZ - 1);
          S[mt][nt][r] += rpe[ridx * NHEAD + h];
        }
      }
      // row max / sum across 4 col-tiles and the 16 lanes of this group
      float mx = -1e30f;
#pragma unroll
      for (int nt = 0; nt < 4; ++nt) mx = fmaxf(mx, S[mt][nt][r]);
#pragma unroll
      for (int off = 8; off > 0; off >>= 1) mx = fmaxf(mx, __shfl_xor(mx, off, 32));
      float sum = 0.f;
#pragma unroll
      for (int nt = 0; nt < 4; ++nt) {
        float e = __expf(S[mt][nt][r] - mx);
        S[mt][nt][r] = e;
        sum += e;
      }
#pragma unroll
      for (int off = 8; off > 0; off >>= 1) sum += __shfl_xor(sum, off, 32);
      float inv = 1.0f / sum;
#pragma unroll
      for (int nt = 0; nt < 4; ++nt) S[mt][nt][r] *= inv;
    }
  }

  // ---- stage P (f16) in LDS to re-fragment C-layout -> A-layout ----
  _Float16* pw = Plds + (size_t)wid * (LP * LP);
#pragma unroll
  for (int mt = 0; mt < 4; ++mt)
#pragma unroll
    for (int nt = 0; nt < 4; ++nt)
#pragma unroll
      for (int r = 0; r < 8; ++r)
        pw[(size_t)(mt * 16 + r + 8 * lg) * LP + nt * 16 + lr] = (_Float16)S[mt][nt][r];

  // ---- O = P v   (K = 64 -> 2 steps, N = 32 -> 2 tiles) ----
  v8f O[4][2];
#pragma unroll
  for (int mt = 0; mt < 4; ++mt) { O[mt][0] = vzero; O[mt][1] = vzero; }

#pragma unroll
  for (int kt = 0; kt < 2; ++kt) {
    v16h vf0 = load_frag16(vt + 0 * 16 * LP + kt * 32, LP);
    v16h vf1 = load_frag16(vt + 1 * 16 * LP + kt * 32, LP);
#pragma unroll
    for (int mt = 0; mt < 4; ++mt) {
      v16h pf = load_frag16(pw + (size_t)(mt * 16) * LP + kt * 32, LP);
      O[mt][0] = wmma16(pf, vf0, O[mt][0]);
      O[mt][1] = wmma16(pf, vf1, O[mt][1]);
    }
  }

  // ---- store merged heads: abuf[w][l][h*32+d] (f16) ----
#pragma unroll
  for (int mt = 0; mt < 4; ++mt)
#pragma unroll
    for (int j = 0; j < 2; ++j) {
      int d = j * 16 + lr;
#pragma unroll
      for (int r = 0; r < 8; ++r) {
        int l = mt * 16 + r + 8 * lg;
        abuf[((size_t)w * LP + l) * CH + h * HD + d] = (_Float16)O[mt][j][r];
      }
    }
}

// --- Kernel 3: output projection fused with un-pad scatter ------------------
// grid (NB, 4), block 256.  Same LDS-staged, pipelined tiling as kernel 1.

__global__ void __launch_bounds__(256)
proj_scatter_kernel(const _Float16* __restrict__ abuf,
                    const _Float16* __restrict__ wout16,
                    const float* __restrict__ bout,
                    float* __restrict__ out) {
  __shared__ _Float16 Bs[2][128 * 32];

  const int w = blockIdx.x;
  int b, qy, qx;
  decode_window(w, b, qy, qx);

  const int wid = threadIdx.x >> 5;
  const int lane = threadIdx.x & 31;
  const int lr = lane & 15, lg = lane >> 4;
  const int m0 = (wid & 3) * 16;
  const int nh = wid >> 2;
  const int nb0 = blockIdx.y * 128;

  const _Float16* arow = abuf + (size_t)w * LP * CH + (size_t)m0 * CH;

  auto stage_B = [&](int kk, int buf) {
#pragma unroll
    for (int c = 0; c < 2; ++c) {
      int ci = c * 256 + threadIdx.x;
      int np = ci >> 2;
      int ho = (ci & 3) * 8;
      copy16_g2l(wout16 + (size_t)(nb0 + np) * CH + kk + ho,
                 &Bs[buf][np * 32 + ho]);
    }
  };

  v8f acc[4];
  const v8f vzero = {0.f,0.f,0.f,0.f,0.f,0.f,0.f,0.f};
#pragma unroll
  for (int j = 0; j < 4; ++j) acc[j] = vzero;

  stage_B(0, 0);
  v16h afCur = load_frag16(arow, CH);
  wait_async();
  __syncthreads();

  int buf = 0;
  for (int kk = 0; kk < CH; kk += 32) {
    const int kn = kk + 32;
    const bool hasNext = kn < CH;
    if (hasNext) stage_B(kn, buf ^ 1);
    v16h afNext = afCur;
    if (hasNext) afNext = load_frag16(arow + kn, CH);

#pragma unroll
    for (int j = 0; j < 4; ++j) {
      v16h bf = load_frag16(&Bs[buf][(nh * 64 + j * 16) * 32], 32);
      acc[j] = wmma16(afCur, bf, acc[j]);
    }

    wait_async();
    __syncthreads();
    afCur = afNext;
    buf ^= 1;
  }

#pragma unroll
  for (int j = 0; j < 4; ++j) {
    int n = nb0 + nh * 64 + j * 16 + lr;
    float bn = bout[n];
#pragma unroll
    for (int r = 0; r < 8; ++r) {
      int l = m0 + r + 8 * lg;
      if (l < LQ) {
        int iy = l / WSZ, ix = l - iy * WSZ;
        int y  = qy * WSZ + iy - PADT;
        int xc = qx * WSZ + ix - PADT;
        if ((unsigned)y < (unsigned)IMG && (unsigned)xc < (unsigned)IMG)
          out[((size_t)b * (IMG * IMG) + (size_t)y * IMG + xc) * CH + n] =
              acc[j][r] + bn;
      }
    }
  }
}

// --- host launcher ----------------------------------------------------------

extern "C" void kernel_launch(void* const* d_in, const int* in_sizes, int n_in,
                              void* d_out, int out_size, void* d_ws, size_t ws_size,
                              hipStream_t stream) {
  const float* x    = (const float*)d_in[0];
  const float* wqkv = (const float*)d_in[1];
  const float* bqkv = (const float*)d_in[2];
  const float* wout = (const float*)d_in[3];
  const float* bout = (const float*)d_in[4];
  const float* rpe  = (const float*)d_in[5];
  float* out = (float*)d_out;

  _Float16* wqkv16 = (_Float16*)d_ws;
  _Float16* wout16 = wqkv16 + N_WQKV;
  _Float16* qbuf   = wout16 + N_WOUT;
  _Float16* kbuf   = qbuf + N_QKV1;
  _Float16* vbufT  = kbuf + N_QKV1;
  _Float16* abuf   = vbufT + N_QKV1;

  cvt_weights_kernel<<<512, 256, 0, stream>>>(wqkv, wout, wqkv16, wout16);

  qkv_gemm_kernel<<<dim3(NB, 12), 256, 0, stream>>>(x, wqkv16, bqkv,
                                                    qbuf, kbuf, vbufT);

  attn_kernel<<<(NB * NHEAD) / 8, 256, 0, stream>>>(qbuf, kbuf, vbufT, rpe, abuf);

  proj_scatter_kernel<<<dim3(NB, 4), 256, 0, stream>>>(abuf, wout16, bout, out);
}
